// BiMambaBlock_2791728742938
// MI455X (gfx1250) — matrix operand order repaired and checked by reading, not compiled, not run on hardware
//
#include <hip/hip_runtime.h>
#include <hip/hip_bf16.h>

// ---------------- problem constants (Mamba2 bi-directional block) ----------
#define D_MODEL   768
#define D_STATE   16
#define HEAD_DIM  64
#define D_INNER   1536            // 2*D_MODEL
#define N_HEADS   24              // D_INNER/HEAD_DIM
#define D_CONV    4
#define CONV_DIM  1568            // D_INNER + 2*D_STATE
#define DIP_W     3128            // 2*D_INNER + 2*D_STATE + N_HEADS
#define SEQ_L     2048
#define BATCH     4
#define M_TOK     (BATCH*SEQ_L)   // 8192
#define HID       3072            // 4*D_MODEL
#define EPSV      1e-5f

// GEMM tiling
#define BM 128
#define BN 128
#define BK 32
#define PITCH 40                  // ushort pitch: 80B rows keep 16B alignment

typedef unsigned short u16;
typedef unsigned long long u64;
typedef __attribute__((ext_vector_type(16))) __bf16 bf16x16;
typedef __attribute__((ext_vector_type(8)))  float  f32x8;

// ---------------- helpers --------------------------------------------------
static __device__ __forceinline__ u16 f32_to_bf16(float f) {
    unsigned int u = __float_as_uint(f);
    unsigned int r = (u + 0x7FFFu + ((u >> 16) & 1u)) >> 16;
    return (u16)r;
}

static __device__ __forceinline__ bf16x16 make_frag(uint4 lo, uint4 hi) {
    union { uint4 u4[2]; bf16x16 v; } c;
    c.u4[0] = lo; c.u4[1] = hi;
    return c.v;
}

static __device__ __forceinline__ void async_b128x2(unsigned lds, u64 gaddr) {
    // copies 32B: LDS[lds + {0,16}] = MEM[gaddr + {0,16}]  (offset applies to both)
    asm volatile("global_load_async_to_lds_b128 %0, %1, off"
                 :: "v"(lds), "v"(gaddr) : "memory");
    asm volatile("global_load_async_to_lds_b128 %0, %1, off offset:16"
                 :: "v"(lds), "v"(gaddr) : "memory");
}

static __device__ __forceinline__ void wait_async0() {
    asm volatile("s_wait_asynccnt 0" ::: "memory");
}

// ---------------- f32 -> bf16 convert --------------------------------------
__global__ __launch_bounds__(256) void cvt_bf16_kernel(const float* __restrict__ in,
                                                       u16* __restrict__ out, int n) {
    int i = blockIdx.x * 256 + threadIdx.x;
    if (i < n) out[i] = f32_to_bf16(in[i]);
}

// ---------------- LayerNorm (f32 in, bf16 out) -----------------------------
__global__ __launch_bounds__(256) void ln_bf16_kernel(const float* __restrict__ x,
                                                      const float* __restrict__ w,
                                                      const float* __restrict__ b,
                                                      u16* __restrict__ out, int D) {
    int row = blockIdx.x, tid = threadIdx.x;
    const float* xr = x + (size_t)row * D;
    __shared__ float red[256];
    float s = 0.f;
    for (int i = tid; i < D; i += 256) s += xr[i];
    red[tid] = s; __syncthreads();
    for (int st = 128; st > 0; st >>= 1) { if (tid < st) red[tid] += red[tid + st]; __syncthreads(); }
    float mean = red[0] / (float)D;
    __syncthreads();
    float v = 0.f;
    for (int i = tid; i < D; i += 256) { float d = xr[i] - mean; v += d * d; }
    red[tid] = v; __syncthreads();
    for (int st = 128; st > 0; st >>= 1) { if (tid < st) red[tid] += red[tid + st]; __syncthreads(); }
    float rstd = rsqrtf(red[0] / (float)D + EPSV);
    u16* orow = out + (size_t)row * D;
    for (int i = tid; i < D; i += 256)
        orow[i] = f32_to_bf16((xr[i] - mean) * rstd * w[i] + b[i]);
}

// ---------------- WMMA bf16 GEMM: D(MxN) = A(MxK) * W(NxK)^T (+bias) -------
// 128x128 block tile, 256 threads = 8 wave32's; each wave: 2x4 = 8 C-frags.
// Double-buffered LDS tiles filled by async global->LDS (ASYNCcnt) copies.
// K % 32 == 0, M % 128 == 0; N guarded (OOB weight rows pre-zeroed in LDS).
__global__ __launch_bounds__(256) void gemm_bf16_kernel(const u16* __restrict__ A,
                                                        const u16* __restrict__ W,
                                                        float* __restrict__ D,
                                                        int M, int N, int K,
                                                        const float* __restrict__ bias) {
    __shared__ u16 As[2][BM][PITCH];
    __shared__ u16 Bs[2][BN][PITCH];
    const int bm = blockIdx.y * BM, bn = blockIdx.x * BN;
    const int tid  = threadIdx.x;
    const int wave = tid >> 5, lane = tid & 31;

    // staging map: thread -> (row, 32B segment); tile = 128 rows x 64B
    const int srow = tid >> 1;
    const int scol = (tid & 1) * 16;             // ushort units
    const u64 aBase = (u64)(A + (size_t)(bm + srow) * K + scol);
    const int wrow  = bn + srow;
    const u64 wBase = (u64)(W + (size_t)wrow * K + scol);
    const bool wOk  = (wrow < N);

    unsigned ldsA[2], ldsB[2];
    ldsA[0] = (unsigned)(size_t)&As[0][srow][scol];
    ldsA[1] = (unsigned)(size_t)&As[1][srow][scol];
    ldsB[0] = (unsigned)(size_t)&Bs[0][srow][scol];
    ldsB[1] = (unsigned)(size_t)&Bs[1][srow][scol];

    if (!wOk) {   // rows beyond N never get async-written: keep them zero forever
        uint4 z = make_uint4(0u, 0u, 0u, 0u);
        *(uint4*)&Bs[0][srow][scol] = z; *(uint4*)&Bs[0][srow][scol + 8] = z;
        *(uint4*)&Bs[1][srow][scol] = z; *(uint4*)&Bs[1][srow][scol + 8] = z;
    }

    auto issue_tile = [&](int k0, int buf) {
        async_b128x2(ldsA[buf], aBase + (u64)k0 * 2u);
        if (wOk) async_b128x2(ldsB[buf], wBase + (u64)k0 * 2u);
    };

    issue_tile(0, 0);
    wait_async0();
    __syncthreads();

    // wave tiling: rows wr..wr+31 (2 frags), cols wc..wc+63 (4 frags)
    const int wr = (wave >> 1) * 32;
    const int wc = (wave & 1) * 64;
    const int am  = lane & 15;
    const int kb  = (lane < 16) ? 0 : 8;     // A-frag K base (ISA 16-bit A layout)
    const int kb2 = (lane < 16) ? 0 : 8;     // B-frag K base

    f32x8 zero8 = {0.f, 0.f, 0.f, 0.f, 0.f, 0.f, 0.f, 0.f};
    f32x8 acc[2][4] = {{zero8, zero8, zero8, zero8}, {zero8, zero8, zero8, zero8}};

    const int nk = K / BK;
    for (int kt = 0; kt < nk; ++kt) {
        const int cur = kt & 1;
        if (kt + 1 < nk) issue_tile((kt + 1) * BK, cur ^ 1);   // overlap copy w/ math

        bf16x16 bfr[4];
        #pragma unroll
        for (int j = 0; j < 4; ++j) {
            const int nr = wc + 16 * j + am;
            bfr[j] = make_frag(*(const uint4*)&Bs[cur][nr][kb2],
                               *(const uint4*)&Bs[cur][nr][kb2 + 16]);
        }
        bf16x16 a0 = make_frag(*(const uint4*)&As[cur][wr + am][kb],
                               *(const uint4*)&As[cur][wr + am][kb + 16]);
        bf16x16 a1 = make_frag(*(const uint4*)&As[cur][wr + 16 + am][kb],
                               *(const uint4*)&As[cur][wr + 16 + am][kb + 16]);
        #pragma unroll
        for (int j = 0; j < 4; ++j)
            acc[0][j] = __builtin_amdgcn_wmma_f32_16x16x32_bf16(false, a0, false, bfr[j],
                                                                (short)0, acc[0][j], false, false);
        #pragma unroll
        for (int j = 0; j < 4; ++j)
            acc[1][j] = __builtin_amdgcn_wmma_f32_16x16x32_bf16(false, a1, false, bfr[j],
                                                                (short)0, acc[1][j], false, false);

        if (kt + 1 < nk) {
            wait_async0();      // this wave's copies into buf^1 have landed
            __syncthreads();    // everyone done reading cur + sees buf^1
        }
    }

    // store (ISA C/D layout): lane&15 = column, lanes>=16 -> rows +8
    const int mofs = (lane < 16) ? 0 : 8;
    #pragma unroll
    for (int i = 0; i < 2; ++i) {
        #pragma unroll
        for (int j = 0; j < 4; ++j) {
            int n = bn + wc + 16 * j + am;
            if (n < N) {
                float bv = bias ? bias[n] : 0.f;
                #pragma unroll
                for (int r = 0; r < 8; ++r)
                    D[(size_t)(bm + wr + 16 * i + mofs + r) * N + n] = acc[i][j][r] + bv;
            }
        }
    }
}

// ---------------- depthwise conv1d + SiLU (blockIdx.y = direction) ---------
__global__ __launch_bounds__(256) void conv_silu_kernel(const float* __restrict__ zx0,
                                                        const float* __restrict__ zx1,
                                                        const float* __restrict__ cw0,
                                                        const float* __restrict__ cw1,
                                                        const float* __restrict__ cb0,
                                                        const float* __restrict__ cb1,
                                                        float* __restrict__ xc0,
                                                        float* __restrict__ xc1) {
    const int dir = blockIdx.y;
    const float* zx    = dir ? zx1 : zx0;
    const float* convw = dir ? cw1 : cw0;
    const float* convb = dir ? cb1 : cb0;
    float* xc          = dir ? xc1 : xc0;
    long idx = (long)blockIdx.x * 256 + threadIdx.x;
    const long total = (long)M_TOK * CONV_DIM;
    if (idx >= total) return;
    int c  = (int)(idx % CONV_DIM);
    long bt = idx / CONV_DIM;
    int t  = (int)(bt % SEQ_L);
    int bb = (int)(bt / SEQ_L);
    float acc = convb[c];
    #pragma unroll
    for (int k = 0; k < D_CONV; ++k) {
        int tt = dir ? (t + (D_CONV - 1) - k) : (t - (D_CONV - 1) + k);
        if (tt >= 0 && tt < SEQ_L)
            acc += convw[k * CONV_DIM + c] *
                   zx[((size_t)bb * SEQ_L + tt) * DIP_W + D_INNER + c];
    }
    acc = acc / (1.f + expf(-acc));          // SiLU
    xc[idx] = acc;
}

// ---------------- selective scan (blockIdx.y = direction) ------------------
// grid (BATCH*N_HEADS, 2); 256 thr: thread holds h[p][ng..ng+3]; quad reduce.
__global__ __launch_bounds__(256) void scan_kernel(const float* __restrict__ zx0,
                                                   const float* __restrict__ zx1,
                                                   const float* __restrict__ xc0,
                                                   const float* __restrict__ xc1,
                                                   const float* __restrict__ dtb0,
                                                   const float* __restrict__ dtb1,
                                                   const float* __restrict__ Alog0,
                                                   const float* __restrict__ Alog1,
                                                   const float* __restrict__ Dp0,
                                                   const float* __restrict__ Dp1,
                                                   float* __restrict__ yr0,
                                                   float* __restrict__ yr1) {
    const int dir = blockIdx.y;
    const float* zx   = dir ? zx1 : zx0;
    const float* xc   = dir ? xc1 : xc0;
    const float* dtb  = dir ? dtb1 : dtb0;
    const float* Alog = dir ? Alog1 : Alog0;
    const float* Dp   = dir ? Dp1 : Dp0;
    float* yraw       = dir ? yr1 : yr0;
    int head = blockIdx.x % N_HEADS;
    int bb   = blockIdx.x / N_HEADS;
    int tid = threadIdx.x;
    int p  = tid >> 2;
    int ng = (tid & 3) * 4;
    float h0 = 0.f, h1 = 0.f, h2 = 0.f, h3 = 0.f;
    float Aneg   = -expf(Alog[head]);
    float dtbias = dtb[head];
    float Dv     = Dp[head];
    for (int s = 0; s < SEQ_L; ++s) {
        int t = dir ? (SEQ_L - 1 - s) : s;
        size_t row = (size_t)bb * SEQ_L + t;
        float draw = zx[row * DIP_W + (D_INNER + CONV_DIM) + head] + dtbias;
        float dtv = (draw > 20.f) ? draw : log1pf(expf(draw));   // softplus
        float dA  = expf(dtv * Aneg);
        const float* xcr = xc + row * CONV_DIM;
        float xv = xcr[head * HEAD_DIM + p];
        float coef = dtv * xv;
        const float* Bp = xcr + D_INNER;
        const float* Cp = xcr + D_INNER + D_STATE;
        float B0 = Bp[ng], B1 = Bp[ng + 1], B2 = Bp[ng + 2], B3 = Bp[ng + 3];
        float C0 = Cp[ng], C1 = Cp[ng + 1], C2 = Cp[ng + 2], C3 = Cp[ng + 3];
        h0 = h0 * dA + coef * B0;
        h1 = h1 * dA + coef * B1;
        h2 = h2 * dA + coef * B2;
        h3 = h3 * dA + coef * B3;
        float yp = h0 * C0 + h1 * C1 + h2 * C2 + h3 * C3;
        yp += __shfl_xor(yp, 1, 32);
        yp += __shfl_xor(yp, 2, 32);
        if ((tid & 3) == 0)
            yraw[row * D_INNER + head * HEAD_DIM + p] = yp + Dv * xv;
    }
}

// ---------------- gated RMSNorm (blockIdx.y = direction) -------------------
__global__ __launch_bounds__(256) void rmsgate_kernel(const float* __restrict__ zx0,
                                                      const float* __restrict__ zx1,
                                                      const float* __restrict__ yr0,
                                                      const float* __restrict__ yr1,
                                                      const float* __restrict__ nw0,
                                                      const float* __restrict__ nw1,
                                                      u16* __restrict__ o0,
                                                      u16* __restrict__ o1) {
    const int dir = blockIdx.y;
    const float* zx    = dir ? zx1 : zx0;
    const float* yraw  = dir ? yr1 : yr0;
    const float* normw = dir ? nw1 : nw0;
    u16* out           = dir ? o1 : o0;
    int row = blockIdx.x, tid = threadIdx.x;
    const float* zr = zx   + (size_t)row * DIP_W;
    const float* yr = yraw + (size_t)row * D_INNER;
    __shared__ float buf[D_INNER];
    __shared__ float red[256];
    float ss = 0.f;
    for (int i = tid; i < D_INNER; i += 256) {
        float z = zr[i];
        float g = yr[i] * (z / (1.f + expf(-z)));
        buf[i] = g; ss += g * g;
    }
    red[tid] = ss; __syncthreads();
    for (int st = 128; st > 0; st >>= 1) { if (tid < st) red[tid] += red[tid + st]; __syncthreads(); }
    float scale = rsqrtf(red[0] / (float)D_INNER + EPSV);
    u16* orow = out + (size_t)row * D_INNER;
    for (int i = tid; i < D_INNER; i += 256)
        orow[i] = f32_to_bf16(buf[i] * scale * normw[i]);
}

// ---------------- small elementwise kernels --------------------------------
__global__ __launch_bounds__(256) void residual3_kernel(const float* __restrict__ a,
                                                        const float* __restrict__ b,
                                                        const float* __restrict__ c,
                                                        float* __restrict__ o, int n) {
    int i = blockIdx.x * 256 + threadIdx.x;
    if (i < n) o[i] = a[i] + b[i] + c[i];
}

__global__ __launch_bounds__(256) void gelu_bf16_kernel(const float* __restrict__ in,
                                                        u16* __restrict__ out, int n) {
    int i = blockIdx.x * 256 + threadIdx.x;
    if (i < n) {
        float v = in[i];
        out[i] = f32_to_bf16(0.5f * v * (1.f + erff(v * 0.70710678118654752f)));
    }
}

__global__ __launch_bounds__(256) void final_add_kernel(const float* __restrict__ xres,
                                                        const float* __restrict__ m,
                                                        const float* __restrict__ bias,
                                                        float* __restrict__ out, int n) {
    int i = blockIdx.x * 256 + threadIdx.x;
    if (i < n) out[i] = xres[i] + m[i] + bias[i % D_MODEL];
}

// ---------------- host-side orchestration ----------------------------------
extern "C" void kernel_launch(void* const* d_in, const int* in_sizes, int n_in,
                              void* d_out, int out_size, void* d_ws, size_t ws_size,
                              hipStream_t stream) {
    (void)in_sizes; (void)n_in; (void)out_size; (void)ws_size;
    const float* x      = (const float*)d_in[0];
    const float* Win[2]   = { (const float*)d_in[1],  (const float*)d_in[9]  };
    const float* convw[2] = { (const float*)d_in[2],  (const float*)d_in[10] };
    const float* convb[2] = { (const float*)d_in[3],  (const float*)d_in[11] };
    const float* dtb[2]   = { (const float*)d_in[4],  (const float*)d_in[12] };
    const float* Alog[2]  = { (const float*)d_in[5],  (const float*)d_in[13] };
    const float* Dp[2]    = { (const float*)d_in[6],  (const float*)d_in[14] };
    const float* normw[2] = { (const float*)d_in[7],  (const float*)d_in[15] };
    const float* Wout[2]  = { (const float*)d_in[8],  (const float*)d_in[16] };
    const float* ln1w = (const float*)d_in[17];
    const float* ln1b = (const float*)d_in[18];
    const float* ln2w = (const float*)d_in[19];
    const float* ln2b = (const float*)d_in[20];
    const float* fc1w = (const float*)d_in[21];
    const float* fc1b = (const float*)d_in[22];
    const float* fc2w = (const float*)d_in[23];
    const float* fc2b = (const float*)d_in[24];
    float* out = (float*)d_out;

    // workspace carve-up (256B aligned)
    char* ws = (char*)d_ws;
    size_t off = 0;
    auto alloc = [&](size_t bytes) -> void* {
        void* p = ws + off;
        off = (off + bytes + 255) & ~(size_t)255;
        return p;
    };
    u16* h16      = (u16*)alloc((size_t)M_TOK * D_MODEL * 2);
    u16* wIn16[2] = { (u16*)alloc((size_t)DIP_W * D_MODEL * 2),
                      (u16*)alloc((size_t)DIP_W * D_MODEL * 2) };
    u16* wOut16[2]= { (u16*)alloc((size_t)D_MODEL * D_INNER * 2),
                      (u16*)alloc((size_t)D_MODEL * D_INNER * 2) };
    u16* wFc116   = (u16*)alloc((size_t)HID * D_MODEL * 2);
    u16* wFc216   = (u16*)alloc((size_t)D_MODEL * HID * 2);
    float* zx[2]  = { (float*)alloc((size_t)M_TOK * DIP_W * 4),
                      (float*)alloc((size_t)M_TOK * DIP_W * 4) };
    float* xc[2]  = { (float*)alloc((size_t)M_TOK * CONV_DIM * 4),
                      (float*)alloc((size_t)M_TOK * CONV_DIM * 4) };
    float* yraw[2]= { (float*)alloc((size_t)M_TOK * D_INNER * 4),
                      (float*)alloc((size_t)M_TOK * D_INNER * 4) };
    u16* yn16[2]  = { (u16*)alloc((size_t)M_TOK * D_INNER * 2),
                      (u16*)alloc((size_t)M_TOK * D_INNER * 2) };
    float* yo[2]  = { (float*)alloc((size_t)M_TOK * D_MODEL * 4),
                      (float*)alloc((size_t)M_TOK * D_MODEL * 4) };
    float* xres   = (float*)alloc((size_t)M_TOK * D_MODEL * 4);
    u16* h2_16    = (u16*)alloc((size_t)M_TOK * D_MODEL * 2);
    // phase-aliased buffers (zx/xc fully consumed before the MLP)
    float* fc1out = zx[0];           // M_TOK*HID floats     <= region
    u16*   m1_16  = (u16*)zx[1];     // M_TOK*HID u16        <= region
    float* fc2out = xc[0];           // M_TOK*D_MODEL floats <= region

    auto ceil_div = [](long a, long b) { return (int)((a + b - 1) / b); };

    // 1) weight conversions to bf16 (stateless, every call)
    cvt_bf16_kernel<<<ceil_div((long)DIP_W * D_MODEL, 256), 256, 0, stream>>>(Win[0], wIn16[0], DIP_W * D_MODEL);
    cvt_bf16_kernel<<<ceil_div((long)DIP_W * D_MODEL, 256), 256, 0, stream>>>(Win[1], wIn16[1], DIP_W * D_MODEL);
    cvt_bf16_kernel<<<ceil_div((long)D_MODEL * D_INNER, 256), 256, 0, stream>>>(Wout[0], wOut16[0], D_MODEL * D_INNER);
    cvt_bf16_kernel<<<ceil_div((long)D_MODEL * D_INNER, 256), 256, 0, stream>>>(Wout[1], wOut16[1], D_MODEL * D_INNER);
    cvt_bf16_kernel<<<ceil_div((long)HID * D_MODEL, 256), 256, 0, stream>>>(fc1w, wFc116, HID * D_MODEL);
    cvt_bf16_kernel<<<ceil_div((long)D_MODEL * HID, 256), 256, 0, stream>>>(fc2w, wFc216, D_MODEL * HID);

    // 2) LN1 -> bf16 tokens
    ln_bf16_kernel<<<M_TOK, 256, 0, stream>>>(x, ln1w, ln1b, h16, D_MODEL);

    // 3) in_proj GEMMs (both directions, unreversed rows)
    {
        dim3 g(ceil_div(DIP_W, BN), M_TOK / BM);
        gemm_bf16_kernel<<<g, 256, 0, stream>>>(h16, wIn16[0], zx[0], M_TOK, DIP_W, D_MODEL, nullptr);
        gemm_bf16_kernel<<<g, 256, 0, stream>>>(h16, wIn16[1], zx[1], M_TOK, DIP_W, D_MODEL, nullptr);
    }

    // 4) conv+SiLU, 5) scan, 6) gated RMSNorm -- both directions fused on grid.y
    {
        dim3 gc(ceil_div((long)M_TOK * CONV_DIM, 256), 2);
        conv_silu_kernel<<<gc, 256, 0, stream>>>(zx[0], zx[1], convw[0], convw[1],
                                                 convb[0], convb[1], xc[0], xc[1]);
        dim3 gs(BATCH * N_HEADS, 2);
        scan_kernel<<<gs, 256, 0, stream>>>(zx[0], zx[1], xc[0], xc[1], dtb[0], dtb[1],
                                            Alog[0], Alog[1], Dp[0], Dp[1], yraw[0], yraw[1]);
        dim3 gr(M_TOK, 2);
        rmsgate_kernel<<<gr, 256, 0, stream>>>(zx[0], zx[1], yraw[0], yraw[1],
                                               normw[0], normw[1], yn16[0], yn16[1]);
    }

    // 7) out_proj GEMMs
    {
        dim3 g(ceil_div(D_MODEL, BN), M_TOK / BM);
        gemm_bf16_kernel<<<g, 256, 0, stream>>>(yn16[0], wOut16[0], yo[0], M_TOK, D_MODEL, D_INNER, nullptr);
        gemm_bf16_kernel<<<g, 256, 0, stream>>>(yn16[1], wOut16[1], yo[1], M_TOK, D_MODEL, D_INNER, nullptr);
    }

    // 8) residual + LN2
    residual3_kernel<<<ceil_div((long)M_TOK * D_MODEL, 256), 256, 0, stream>>>(x, yo[0], yo[1], xres, M_TOK * D_MODEL);
    ln_bf16_kernel<<<M_TOK, 256, 0, stream>>>(xres, ln2w, ln2b, h2_16, D_MODEL);

    // 9) MLP: fc1 (+bias) -> GELU -> fc2 -> final residual (+fc2 bias)
    {
        dim3 g1(ceil_div(HID, BN), M_TOK / BM);
        gemm_bf16_kernel<<<g1, 256, 0, stream>>>(h2_16, wFc116, fc1out, M_TOK, HID, D_MODEL, fc1b);
        gelu_bf16_kernel<<<ceil_div((long)M_TOK * HID, 256), 256, 0, stream>>>(fc1out, m1_16, M_TOK * HID);
        dim3 g2(ceil_div(D_MODEL, BN), M_TOK / BM);
        gemm_bf16_kernel<<<g2, 256, 0, stream>>>(m1_16, wFc216, fc2out, M_TOK, D_MODEL, HID, nullptr);
        final_add_kernel<<<ceil_div((long)M_TOK * D_MODEL, 256), 256, 0, stream>>>(xres, fc2out, fc2b, out, M_TOK * D_MODEL);
    }
}
